// Encoder_11373073400145
// MI455X (gfx1250) — matrix operand order repaired and checked by reading, not compiled, not run on hardware
//
#include <hip/hip_runtime.h>
#include <hip/hip_bf16.h>

// ---------------------------------------------------------------------------
// PCGCv2-style masked 3D-CNN encoder for MI455X (gfx1250, wave32, WMMA).
//
// Roofline: ~70 G MACs, <140 MB live working set -> everything is L2-resident
// (192 MB L2, 23.3 TB/s HBM), so the kernel is built around:
//  * v_wmma_f32_16x16x32_f16 implicit GEMM (f16 in, f32 accumulate)
//  * f16 CHANNEL-LAST activations [B][vox][C]: every global/LDS access is a
//    contiguous b128 op (staging, epilogue store, residual read), including
//    the stride-2 down-convs.
//  * fragment-major pre-packed f16 weights: a 16x32 A fragment is two
//    coalesced global_load_b128s, zero per-tap ALU.
//  * N-tiling (NV voxel subtiles per wave): one A fragment feeds NV WMMAs on
//    NV independent accumulators -> NV x arithmetic intensity per weight
//    fetch, 1/NV barriers per WMMA, no back-to-back same-accumulator WMMA.
//  * double-buffered LDS B tile: one barrier per tap, next tap's b128 loads
//    overlap the current tap's WMMAs.
// ---------------------------------------------------------------------------

typedef __attribute__((ext_vector_type(16))) _Float16 v16h;
typedef __attribute__((ext_vector_type(8)))  _Float16 v8h;
typedef __attribute__((ext_vector_type(8)))  float    v8f;

constexpr int BATCH = 8;

constexpr int ilog2c(int v) { return v <= 1 ? 0 : 1 + ilog2c(v >> 1); }

// ---------------------------------------------------------------------------
// 2x2x2 max-pool of the occupancy map (values are {0,1}), f32 in/out.
// ---------------------------------------------------------------------------
__global__ void maxpool2_kernel(const float* __restrict__ in,
                                float* __restrict__ out,
                                int Din, int Dout, int total) {
  int idx = blockIdx.x * blockDim.x + threadIdx.x;
  if (idx >= total) return;
  int voxN = Dout * Dout * Dout;
  int bb  = idx / voxN;
  int vox = idx - bb * voxN;
  int x0 = vox % Dout;
  int y0 = (vox / Dout) % Dout;
  int z0 = vox / (Dout * Dout);
  size_t base = (size_t)bb * Din * Din * Din;
  float m = 0.0f;
  for (int dz = 0; dz < 2; ++dz)
    for (int dy = 0; dy < 2; ++dy)
      for (int dx = 0; dx < 2; ++dx) {
        float v = in[base + ((size_t)(2 * z0 + dz) * Din + (2 * y0 + dy)) * Din
                          + (2 * x0 + dx)];
        m = fmaxf(m, v);
      }
  out[idx] = m;
}

// ---------------------------------------------------------------------------
// Weight repack: f32 [coutReal][cin][taps] -> f16 fragment-major layout:
//   wf[(((cg*taps + tap)*kg + g)*32 + lane)*16 + j]
// with the contiguous 16-bit A layout: m = lane&15, k = g*32 + j +
// (lane>=16?16:0).  Out-of-range cout/cin zero-padded.
// ---------------------------------------------------------------------------
__global__ void repack_w_kernel(const float* __restrict__ w,
                                _Float16* __restrict__ wf,
                                int cin, int taps, int kg, int coutReal,
                                int total) {
  int idx = blockIdx.x * blockDim.x + threadIdx.x;
  if (idx >= total) return;
  int j = idx & 15;
  int l = (idx >> 4) & 31;
  int rest = idx >> 9;
  int g = rest % kg;   rest /= kg;
  int tap = rest % taps;
  int cg = rest / taps;
  int k  = g * 32 + j + ((l >= 16) ? 16 : 0);
  int co = cg * 16 + (l & 15);
  float v = 0.0f;
  if (co < coutReal && k < cin) v = w[((size_t)co * cin + k) * taps + tap];
  wf[idx] = (_Float16)v;
}

// ---------------------------------------------------------------------------
// conv0: Cin=1 -> 16, 3x3x3 SAME, h0 = relu((conv+b)*m0), output f16
// channel-last.  K=1 makes WMMA pointless; VALU + ~10% sparsity early-out.
// ---------------------------------------------------------------------------
__global__ void conv0_kernel(const float* __restrict__ x,
                             const float* __restrict__ w,    // [16][27]
                             const float* __restrict__ bias, // [16]
                             _Float16* __restrict__ out) {   // [B][64^3][16]
  constexpr int D = 64, VOX = D * D * D, CO = 16;
  __shared__ float ws[CO * 27 + CO];
  for (int i = threadIdx.x; i < CO * 27 + CO; i += blockDim.x)
    ws[i] = (i < CO * 27) ? w[i] : bias[i - CO * 27];
  __syncthreads();

  int idx = blockIdx.x * blockDim.x + threadIdx.x;
  if (idx >= BATCH * VOX) return;
  int bb = idx / VOX, vox = idx - bb * VOX;

  float m0 = x[idx];
  float acc[CO];
#pragma unroll
  for (int co = 0; co < CO; ++co) acc[co] = 0.0f;

  if (m0 != 0.0f) {
    int x0 = vox & 63, y0 = (vox >> 6) & 63, z0 = vox >> 12;
    const float* xb = x + (size_t)bb * VOX;
    for (int tap = 0; tap < 27; ++tap) {
      int dz = tap / 9 - 1, dy = (tap / 3) % 3 - 1, dx = tap % 3 - 1;
      int zz = z0 + dz, yy = y0 + dy, xx = x0 + dx;
      if ((unsigned)zz < 64u && (unsigned)yy < 64u && (unsigned)xx < 64u) {
        float v = xb[(zz << 12) + (yy << 6) + xx];
        if (v != 0.0f) {
#pragma unroll
          for (int co = 0; co < CO; ++co) acc[co] += ws[co * 27 + tap] * v;
        }
      }
    }
  }
  v8h lo, hi;
#pragma unroll
  for (int co = 0; co < 8; ++co) {
    float t0 = (m0 != 0.0f) ? fmaxf(acc[co] + ws[CO * 27 + co], 0.0f) : 0.0f;
    float t1 = (m0 != 0.0f) ? fmaxf(acc[co + 8] + ws[CO * 27 + co + 8], 0.0f) : 0.0f;
    lo[co] = (_Float16)t0;
    hi[co] = (_Float16)t1;
  }
  size_t base = ((size_t)bb * VOX + vox) * CO;
  *(v8h*)(out + base)     = lo;
  *(v8h*)(out + base + 8) = hi;
}

// ---------------------------------------------------------------------------
// Generic masked conv as implicit GEMM on v_wmma_f32_16x16x32_f16.
//   KS==3 : 3x3x3 SAME (27 taps, bounds-guarded)
//   KS==2 : 2x2x2 stride-2 VALID (8 taps, always in bounds)
// Each block owns 16*NV output voxels; wave cg owns couts [cg*16, cg*16+16)
// for all NV subtiles (NV independent accumulators share each A fragment).
// ---------------------------------------------------------------------------
template <int CIN, int COUT, int KS, int DOUT, int NV>
__global__ __launch_bounds__(32 * (COUT / 16)) void wmma_conv_kernel(
    const _Float16* __restrict__ in,   // [B][DIN^3][CIN]
    const _Float16* __restrict__ wf,   // fragment-major packed weights
    const float* __restrict__ bias,    // [coutReal]
    const float* __restrict__ mask,    // [B][DOUT^3] f32
    const _Float16* __restrict__ res,  // [B][DOUT^3][COUT] or nullptr
    _Float16* __restrict__ out,        // [B][DOUT^3][COUT]
    float* __restrict__ out32,         // optional NCDHW f32 (final layer)
    int relu, int coutReal) {
  constexpr int CINP  = ((CIN + 31) / 32) * 32;
  constexpr int KG    = CINP / 32;
  constexpr int TAPS  = KS * KS * KS;
  constexpr int DIN   = (KS == 2) ? 2 * DOUT : DOUT;
  constexpr int LOG   = ilog2c(DOUT);
  constexpr int VOXN  = DOUT * DOUT * DOUT;
  constexpr size_t DIN3 = (size_t)DIN * DIN * DIN;
  constexpr int NT    = 32 * (COUT / 16);
  constexpr int NVOX  = 16 * NV;            // voxels per block
  constexpr int TILES = VOXN / NVOX;
  constexpr int CROWS = CINP / 8;           // 8-half chunks per voxel
  constexpr int CHUNKS = NVOX * CROWS;

  __shared__ __align__(32) _Float16 bst[2][NVOX][CINP];  // double-buffered

  const int tid  = threadIdx.x;
  const int lane = tid & 31;
  const int cg   = tid >> 5;
  const int bb   = blockIdx.x / TILES;
  const int vb   = (blockIdx.x % TILES) * NVOX;

  // Stage the CIN x NVOX im2col patch for one tap into LDS buffer p:
  // one b128 global load + one b128 LDS store per chunk.
  auto stage = [&](int tap, int p) {
    int dz, dy, dx;
    if constexpr (KS == 3) {
      dz = tap / 9 - 1; dy = (tap / 3) % 3 - 1; dx = tap % 3 - 1;
    } else {
      dz = tap >> 2; dy = (tap >> 1) & 1; dx = tap & 1;
    }
    for (int ch = tid; ch < CHUNKS; ch += NT) {
      int n  = ch / CROWS;
      int kc = (ch - n * CROWS) * 8;
      int vox = vb + n;
      int x0 = vox & (DOUT - 1);
      int y0 = (vox >> LOG) & (DOUT - 1);
      int z0 = vox >> (2 * LOG);
      int zz, yy, xx;
      bool inb;
      if constexpr (KS == 3) {
        zz = z0 + dz; yy = y0 + dy; xx = x0 + dx;
        inb = ((unsigned)zz < (unsigned)DIN) & ((unsigned)yy < (unsigned)DIN) &
              ((unsigned)xx < (unsigned)DIN);
      } else {
        zz = 2 * z0 + dz; yy = 2 * y0 + dy; xx = 2 * x0 + dx;
        inb = true;
      }
      v8h v = {};
      if (inb && kc < CIN)
        v = *(const v8h*)(in +
              ((size_t)bb * DIN3 + ((size_t)zz * DIN + yy) * DIN + xx) * CIN + kc);
      *(v8h*)&bst[p][n][kc] = v;
    }
  };

  v8f c[NV];
#pragma unroll
  for (int v = 0; v < NV; ++v) c[v] = (v8f){};
  const int n  = lane & 15;             // voxel column within a subtile (N)
  const int hi = (lane >= 16) ? 1 : 0;  // lane-half selector

  stage(0, 0);
  for (int tap = 0; tap < TAPS; ++tap) {
    __syncthreads();
    if (tap + 1 < TAPS) stage(tap + 1, (tap + 1) & 1);  // overlap next tap
    const int p = tap & 1;
#pragma unroll
    for (int g = 0; g < KG; ++g) {
      // One A fragment (2x global_load_b128) feeds NV WMMAs.
      v16h afrag = *(const v16h*)(wf +
          ((((size_t)cg * TAPS + tap) * KG + g) << 9) + (lane << 4));
#pragma unroll
      for (int v = 0; v < NV; ++v) {
        v16h bfrag = *(const v16h*)&bst[p][16 * v + n][g * 32 + hi * 16];
        c[v] = __builtin_amdgcn_wmma_f32_16x16x32_f16(
            /*neg_a=*/false, afrag, /*neg_b=*/false, bfrag,
            /*c_mod=*/(short)0, c[v], /*reuse_a=*/false, /*reuse_b=*/false);
      }
    }
  }

#pragma unroll
  for (int v = 0; v < NV; ++v) {
    const int vox = vb + 16 * v + n;
    const float mv = mask[(size_t)bb * VOXN + vox];
    if (out32) {
      // Final layer: scattered f32 NCDHW store straight into d_out.
#pragma unroll
      for (int r = 0; r < 8; ++r) {
        int co = cg * 16 + hi * 8 + r;
        if (co < coutReal) {
          float t = c[v][r] + bias[co];
          if (relu) t = fmaxf(t, 0.0f);
          t *= mv;
          out32[((size_t)bb * coutReal + co) * VOXN + vox] = t;
        }
      }
    } else {
      size_t base = ((size_t)bb * VOXN + vox) * COUT + cg * 16 + hi * 8;
      v8h rv = {};
      if (res) rv = *(const v8h*)(res + base);
      v8h hv;
#pragma unroll
      for (int r = 0; r < 8; ++r) {
        float t = c[v][r] + bias[cg * 16 + hi * 8 + r];
        if (relu) t = fmaxf(t, 0.0f);
        t *= mv;
        t += (float)rv[r];
        hv[r] = (_Float16)t;
      }
      *(v8h*)(out + base) = hv;  // one packed b128 store per lane
    }
  }
}

// ---------------------------------------------------------------------------
// f16 channel-last -> f32 NCDHW (for the out0/out1 slices of d_out).
// ---------------------------------------------------------------------------
__global__ void chlast_to_nchw_f32(const _Float16* __restrict__ in,
                                   float* __restrict__ out,
                                   int C, int voxN, int total) {
  int idx = blockIdx.x * blockDim.x + threadIdx.x;
  if (idx >= total) return;
  int bb  = idx / (C * voxN);
  int rem = idx - bb * C * voxN;
  int co  = rem / voxN;
  int vox = rem - co * voxN;
  out[idx] = (float)in[((size_t)bb * voxN + vox) * C + co];
}

// ---------------------------------------------------------------------------
// Host side: resolve flattened-pytree input pointers, run the pipeline.
// ---------------------------------------------------------------------------
extern "C" void kernel_launch(void* const* d_in, const int* in_sizes, int n_in,
                              void* d_out, int out_size, void* d_ws,
                              size_t ws_size, hipStream_t stream) {
  (void)n_in; (void)out_size; (void)ws_size;
  auto G = [&](int i) { return (const float*)d_in[i]; };

  const float *xin;
  const float *c0w, *c0b, *d0w, *d0b, *c1w, *c1b, *d1w, *d1b;
  const float *c2w, *c2b, *d2w, *d2b, *c3w, *c3b;
  const float *b0[3][4], *b1[3][4], *b2[3][4];

  // Two plausible flattenings of setup_inputs():
  //  (a) jax pytree order (sorted dict keys): mask, params{block0..down2}, x
  //  (b) dict insertion order: params{conv0,down0,block0,...}, x, mask
  if (in_sizes[0] == BATCH * 64 * 64 * 64) {
    int i = 1;  // skip mask; x (== mask as float) is used instead
    for (int t = 0; t < 3; ++t) for (int j = 0; j < 4; ++j) b0[t][j] = G(i++);
    for (int t = 0; t < 3; ++t) for (int j = 0; j < 4; ++j) b1[t][j] = G(i++);
    for (int t = 0; t < 3; ++t) for (int j = 0; j < 4; ++j) b2[t][j] = G(i++);
    c0w = G(i++); c0b = G(i++);
    c1w = G(i++); c1b = G(i++);
    c2w = G(i++); c2b = G(i++);
    c3w = G(i++); c3b = G(i++);
    d0w = G(i++); d0b = G(i++);
    d1w = G(i++); d1b = G(i++);
    d2w = G(i++); d2b = G(i++);
    xin = G(i++);
  } else {
    int i = 0;
    c0w = G(i++); c0b = G(i++);
    d0w = G(i++); d0b = G(i++);
    for (int t = 0; t < 3; ++t) for (int j = 0; j < 4; ++j) b0[t][j] = G(i++);
    c1w = G(i++); c1b = G(i++);
    d1w = G(i++); d1b = G(i++);
    for (int t = 0; t < 3; ++t) for (int j = 0; j < 4; ++j) b1[t][j] = G(i++);
    c2w = G(i++); c2b = G(i++);
    d2w = G(i++); d2b = G(i++);
    for (int t = 0; t < 3; ++t) for (int j = 0; j < 4; ++j) b2[t][j] = G(i++);
    c3w = G(i++); c3b = G(i++);
    xin = G(i++);
  }

  // Workspace allocator (256B-aligned carve-outs), ~134 MB total.
  char* wsb = (char*)d_ws;
  size_t off = 0;
  auto alloc = [&](size_t bytes) {
    void* p = wsb + off;
    off = (off + bytes + 255) & ~(size_t)255;
    return p;
  };

  float* m1 = (float*)alloc((size_t)BATCH * 32768 * 4);
  float* m2 = (float*)alloc((size_t)BATCH * 4096 * 4);
  float* m3 = (float*)alloc((size_t)BATCH * 512 * 4);
  _Float16* h0 = (_Float16*)alloc((size_t)BATCH * 262144 * 16 * 2);
  _Float16* a0 = (_Float16*)alloc((size_t)BATCH * 32768 * 32 * 2);
  _Float16* t0 = (_Float16*)alloc((size_t)BATCH * 32768 * 32 * 2);
  _Float16* h1 = (_Float16*)alloc((size_t)BATCH * 32768 * 32 * 2);
  _Float16* a1 = (_Float16*)alloc((size_t)BATCH * 4096 * 64 * 2);
  _Float16* t1 = (_Float16*)alloc((size_t)BATCH * 4096 * 64 * 2);
  _Float16* h2 = (_Float16*)alloc((size_t)BATCH * 4096 * 64 * 2);
  _Float16* a2 = (_Float16*)alloc((size_t)BATCH * 512 * 32 * 2);
  _Float16* t2 = (_Float16*)alloc((size_t)BATCH * 512 * 32 * 2);

  // One-time (per launch) weight repacks -> fragment-major f16, L2-resident.
  auto mkw = [&](const float* w, int coutp, int cin, int taps, int coutReal) {
    int cinp = ((cin + 31) / 32) * 32, kg = cinp / 32;
    int total = (coutp / 16) * taps * kg * 512;
    _Float16* wf = (_Float16*)alloc((size_t)total * 2);
    repack_w_kernel<<<(total + 255) / 256, 256, 0, stream>>>(w, wf, cin, taps,
                                                             kg, coutReal, total);
    return wf;
  };
  _Float16* Wd0 = mkw(d0w, 32, 16, 8, 32);
  _Float16* Wb0[3][2];
  for (int t = 0; t < 3; ++t) {
    Wb0[t][0] = mkw(b0[t][0], 32, 32, 27, 32);
    Wb0[t][1] = mkw(b0[t][2], 32, 32, 27, 32);
  }
  _Float16* Wc1 = mkw(c1w, 32, 32, 27, 32);
  _Float16* Wd1 = mkw(d1w, 64, 32, 8, 64);
  _Float16* Wb1[3][2];
  for (int t = 0; t < 3; ++t) {
    Wb1[t][0] = mkw(b1[t][0], 64, 64, 27, 64);
    Wb1[t][1] = mkw(b1[t][2], 64, 64, 27, 64);
  }
  _Float16* Wc2 = mkw(c2w, 64, 64, 27, 64);
  _Float16* Wd2 = mkw(d2w, 32, 64, 8, 32);
  _Float16* Wb2[3][2];
  for (int t = 0; t < 3; ++t) {
    Wb2[t][0] = mkw(b2[t][0], 32, 32, 27, 32);
    Wb2[t][1] = mkw(b2[t][2], 32, 32, 27, 32);
  }
  _Float16* Wc3 = mkw(c3w, 16, 32, 27, 8);

  // Mask pyramid (m0 == xin).
  maxpool2_kernel<<<(BATCH * 32768 + 255) / 256, 256, 0, stream>>>(xin, m1, 64, 32, BATCH * 32768);
  maxpool2_kernel<<<(BATCH * 4096 + 255) / 256, 256, 0, stream>>>(m1, m2, 32, 16, BATCH * 4096);
  maxpool2_kernel<<<(BATCH * 512 + 255) / 256, 256, 0, stream>>>(m2, m3, 16, 8, BATCH * 512);

  // conv0 (1->16, 64^3) -> h0 (f16 channel-last).
  conv0_kernel<<<(BATCH * 262144) / 256, 256, 0, stream>>>(xin, c0w, c0b, h0);

  // down0 (16->32, 64^3 -> 32^3) + block0 + conv1.   NV=4 -> 4096 blocks.
  wmma_conv_kernel<16, 32, 2, 32, 4><<<BATCH * 512, 64, 0, stream>>>(h0, Wd0, d0b, m1, nullptr, a0, nullptr, 1, 32);
  for (int t = 0; t < 3; ++t) {
    wmma_conv_kernel<32, 32, 3, 32, 4><<<BATCH * 512, 64, 0, stream>>>(a0, Wb0[t][0], b0[t][1], m1, nullptr, t0, nullptr, 1, 32);
    wmma_conv_kernel<32, 32, 3, 32, 4><<<BATCH * 512, 64, 0, stream>>>(t0, Wb0[t][1], b0[t][3], m1, a0, a0, nullptr, 0, 32);
  }
  wmma_conv_kernel<32, 32, 3, 32, 4><<<BATCH * 512, 64, 0, stream>>>(a0, Wc1, c1b, m1, nullptr, h1, nullptr, 1, 32);

  // down1 (32->64, 32^3 -> 16^3) + block1 + conv2.   NV=4 -> 512 blocks.
  wmma_conv_kernel<32, 64, 2, 16, 4><<<BATCH * 64, 128, 0, stream>>>(h1, Wd1, d1b, m2, nullptr, a1, nullptr, 1, 64);
  for (int t = 0; t < 3; ++t) {
    wmma_conv_kernel<64, 64, 3, 16, 4><<<BATCH * 64, 128, 0, stream>>>(a1, Wb1[t][0], b1[t][1], m2, nullptr, t1, nullptr, 1, 64);
    wmma_conv_kernel<64, 64, 3, 16, 4><<<BATCH * 64, 128, 0, stream>>>(t1, Wb1[t][1], b1[t][3], m2, a1, a1, nullptr, 0, 64);
  }
  wmma_conv_kernel<64, 64, 3, 16, 4><<<BATCH * 64, 128, 0, stream>>>(a1, Wc2, c2b, m2, nullptr, h2, nullptr, 1, 64);

  // down2 (64->32, 16^3 -> 8^3) + block2.   NV=2 -> 128 blocks.
  wmma_conv_kernel<64, 32, 2, 8, 2><<<BATCH * 16, 64, 0, stream>>>(h2, Wd2, d2b, m3, nullptr, a2, nullptr, 1, 32);
  for (int t = 0; t < 3; ++t) {
    wmma_conv_kernel<32, 32, 3, 8, 2><<<BATCH * 16, 64, 0, stream>>>(a2, Wb2[t][0], b2[t][1], m3, nullptr, t2, nullptr, 1, 32);
    wmma_conv_kernel<32, 32, 3, 8, 2><<<BATCH * 16, 64, 0, stream>>>(t2, Wb2[t][1], b2[t][3], m3, a2, a2, nullptr, 0, 32);
  }

  // conv3 (32->8, 8^3), Cout padded to 16; writes out2 (f32 NCDHW) to d_out.
  float* outp = (float*)d_out;
  wmma_conv_kernel<32, 16, 3, 8, 2><<<BATCH * 16, 32, 0, stream>>>(a2, Wc3, c3b, m3, nullptr, nullptr, outp, 0, 8);

  // d_out = [out2 | out1 | out0] flat, f32 NCDHW.
  chlast_to_nchw_f32<<<(2097152 + 255) / 256, 256, 0, stream>>>(a1, outp + 32768, 64, 4096, 2097152);
  chlast_to_nchw_f32<<<(8388608 + 255) / 256, 256, 0, stream>>>(a0, outp + 32768 + 2097152, 32, 32768, 8388608);
}
